// GCN_GAT_83038897701523
// MI455X (gfx1250) — matrix operand order repaired and checked by reading, not compiled, not run on hardware
//
#include <hip/hip_runtime.h>
#include <hip/hip_bf16.h>
#include <stdint.h>

// ---------------------------------------------------------------------------
// Types for CDNA5 WMMA (gfx1250, wave32)
// ---------------------------------------------------------------------------
typedef __attribute__((ext_vector_type(16))) __bf16 v16bf;
typedef __attribute__((ext_vector_type(8)))  float  v8f;

union Frag16 {
  v16bf v;
  uint4 u[2];
};

#define NEG_SLOPE 0.2f
#define EPSF 1e-16f

static __device__ __forceinline__ unsigned short f2bf(float f) {
  unsigned int u = __float_as_uint(f);
  u += 0x7FFFu + ((u >> 16) & 1u);   // round-to-nearest-even
  return (unsigned short)(u >> 16);
}

// monotonic uint key for float atomicMax
static __device__ __forceinline__ unsigned int fkey(float f) {
  unsigned int u = __float_as_uint(f);
  return (u & 0x80000000u) ? ~u : (u | 0x80000000u);
}
static __device__ __forceinline__ float funkey(unsigned int k) {
  unsigned int u = (k & 0x80000000u) ? (k & 0x7FFFFFFFu) : ~k;
  return __uint_as_float(u);
}

static __device__ __forceinline__ int esrc(const int* ei, int e, int E) {
  return (e < E) ? ei[e] : (e - E);
}
static __device__ __forceinline__ int edst(const int* ei, int e, int E) {
  return (e < E) ? ei[E + e] : (e - E);
}

// ---- CDNA5 async copy: global -> LDS, tracked by ASYNCcnt ----
static __device__ __forceinline__ void async_copy_b128(unsigned lds_byte_addr,
                                                       const void* gsrc) {
  asm volatile("global_load_async_to_lds_b128 %0, %1, off"
               :: "v"(lds_byte_addr), "v"(gsrc) : "memory");
}
static __device__ __forceinline__ void wait_async0() {
  asm volatile("s_wait_asynccnt 0x0" ::: "memory");
}

// ---------------------------------------------------------------------------
// Utility kernels
// ---------------------------------------------------------------------------
__global__ void k_zero_f32(float* p, long n) {
  long i = (long)blockIdx.x * blockDim.x + threadIdx.x;
  if (i < n) p[i] = 0.0f;
}
__global__ void k_zero_u32(unsigned int* p, long n) {
  long i = (long)blockIdx.x * blockDim.x + threadIdx.x;
  if (i < n) p[i] = 0u;
}
__global__ void k_f32_to_bf16(const float* __restrict__ in,
                              unsigned short* __restrict__ out, long n) {
  long i = (long)blockIdx.x * blockDim.x + threadIdx.x;
  if (i < n) out[i] = f2bf(in[i]);
}

// ---------------------------------------------------------------------------
// WMMA GEMM:  C(MxNc) = A(MxK, bf16 row-major) * B(KxNc, bf16 row-major) + bias
// Block: 256 threads = 8 waves. Block tile 128x128, wave tile 16x128
// (8 accumulators). B tile (32x128 per K-step) streamed into double-buffered
// LDS with global_load_async_to_lds_b128 (ASYNCcnt), overlapping the WMMAs
// of the current tile. All 8 B fragments are fetched from LDS before the WMMA
// burst so the XDL ops can issue back-to-back behind one s_wait_dscnt.
// Requires: K % 32 == 0, Nc % 128 == 0. M guarded.
// ---------------------------------------------------------------------------
#define BSTRIDE 136   // ushorts per LDS row: 272 B (16B-aligned, bank-skewed)

__global__ __launch_bounds__(256, 1) void k_wmma_gemm(
    const unsigned short* __restrict__ A,
    const unsigned short* __restrict__ B,
    const float* __restrict__ bias,
    float* __restrict__ C,
    int M, int K, int Nc, int relu)
{
  __shared__ __align__(16) unsigned short bsh[2 * 32 * BSTRIDE];

  const int tid  = threadIdx.x;
  const int lane = tid & 31;
  const int wave = tid >> 5;
  const int tm0  = blockIdx.x * 128 + wave * 16;
  const int tn0  = blockIdx.y * 128;

  v8f acc[8] = {};

  // A fragment addressing (16-bit A 16x32 layout):
  // lane L<16 holds row tm0+L, K chunks [0..7] and [16..23];
  // lane L+16 holds same row, chunks [8..15] and [24..31].
  int arow = tm0 + (lane & 15);
  if (arow > M - 1) arow = M - 1;
  const int koff = (lane >> 4) << 3;            // 0 or 8
  const unsigned short* aptr = A + (size_t)arow * K + koff;

  // async staging geometry: thread t moves 32 B (two b128 chunks) of the
  // 32x128 B tile per K-step. Stepping pointer avoids per-iter 64-bit mul.
  const int srow = tid >> 3;                    // 0..31
  const int scol = (tid & 7) << 4;              // 0,16,...,112
  const unsigned short* srcp = B + (size_t)srow * Nc + tn0 + scol;
  const size_t sstep = (size_t)32 * Nc;         // elements per K-step

  // LDS byte address of the staging buffer (addrspace(3) offset)
  const unsigned lbase =
      (unsigned)(size_t)((__attribute__((address_space(3))) unsigned short*)bsh);
  const unsigned dst_row_off = (unsigned)(srow * BSTRIDE + scol) * 2u;

  // prologue: stage tile k0=0 into buffer 0
  async_copy_b128(lbase + dst_row_off,       (const void*)(srcp));
  async_copy_b128(lbase + dst_row_off + 16u, (const void*)(srcp + 8));
  srcp += sstep;

  int buf = 0;
  for (int k0 = 0; k0 < K; k0 += 32) {
    wait_async0();            // this wave's copies for tile k0 have landed
    __syncthreads();          // every wave's copies landed; prev reads done

    if (k0 + 32 < K) {        // stage tile k0+32 into the other buffer
      const unsigned dsto =
          lbase + (unsigned)((buf ^ 1) * 32 * BSTRIDE * 2) + dst_row_off;
      async_copy_b128(dsto,       (const void*)(srcp));
      async_copy_b128(dsto + 16u, (const void*)(srcp + 8));
      srcp += sstep;
    }

    // A fragment for this K-step
    Frag16 a;
    const unsigned short* ap = aptr + k0;
    a.u[0] = *(const uint4*)(ap);
    a.u[1] = *(const uint4*)(ap + 16);
    __builtin_prefetch(ap + 128, 0, 3);         // global_prefetch_b8 (WGP)

    // B fragments: lane l holds col (l&15) per tile, K rows 8h+[0..7] and 16+8h+[0..7].
    // Fetch all 8 fragments first, then burst the 8 WMMAs.
    const unsigned short* bl = bsh + (size_t)buf * 32 * BSTRIDE + ((lane >> 4) << 3) * BSTRIDE + (lane & 15);
    Frag16 bf[8];
#pragma unroll
    for (int t = 0; t < 8; ++t) {
#pragma unroll
      for (int s = 0; s < 16; ++s)
        bf[t].v[s] = __builtin_bit_cast(__bf16, bl[(s < 8 ? s : s + 8) * BSTRIDE + t * 16]);
    }
#pragma unroll
    for (int t = 0; t < 8; ++t) {
      acc[t] = __builtin_amdgcn_wmma_f32_16x16x32_bf16(
          false, a.v, false, bf[t].v, (short)0, acc[t], false, false);
    }
    buf ^= 1;
  }

  // C/D layout: VGPR r -> M = r (lanes 0-15) or 8+r (lanes 16-31); N = lane&15
  const int rbase = tm0 + ((lane >> 4) << 3);
  const int cbase = tn0 + (lane & 15);
#pragma unroll
  for (int t = 0; t < 8; ++t) {
    const int cn = cbase + t * 16;
    const float bb = bias ? bias[cn] : 0.0f;
#pragma unroll
    for (int r = 0; r < 8; ++r) {
      const int row = rbase + r;
      if (row < M) {
        float v = acc[t][r] + bb;
        if (relu) v = fmaxf(v, 0.0f);
        C[(size_t)row * Nc + cn] = v;
      }
    }
  }
}

// ---------------------------------------------------------------------------
// Edge phase kernels (GATv2 attention). One wave per (edge, head).
// ---------------------------------------------------------------------------
__global__ __launch_bounds__(256) void k_edge_score(
    const int* __restrict__ ei, const float* __restrict__ xl,
    const float* __restrict__ xr, const float* __restrict__ att,
    float* __restrict__ s, unsigned int* __restrict__ mkey,
    int E, int Etot, int HH, int C)
{
  const int task = blockIdx.x * (blockDim.x >> 5) + (threadIdx.x >> 5);
  if (task >= Etot * HH) return;
  const int e = task / HH, h = task % HH;
  const int lane = threadIdx.x & 31;
  const int sn = esrc(ei, e, E), dn = edst(ei, e, E);
  const float* pl = xl + ((size_t)sn * HH + h) * C;
  const float* pr = xr + ((size_t)dn * HH + h) * C;
  const float* pa = att + (size_t)h * C;
  float acc = 0.0f;
  for (int c = lane; c < C; c += 32) {
    float v = pl[c] + pr[c];
    v = (v > 0.0f) ? v : v * NEG_SLOPE;
    acc += v * pa[c];
  }
  for (int off = 16; off > 0; off >>= 1) acc += __shfl_down(acc, off, 32);
  if (lane == 0) {
    s[task] = acc;
    atomicMax(&mkey[dn * HH + h], fkey(acc));
  }
}

__global__ void k_edge_exp(
    const int* __restrict__ ei, const float* __restrict__ s,
    const unsigned int* __restrict__ mkey, float* __restrict__ ex,
    float* __restrict__ den, int E, int Etot, int HH)
{
  const int i = blockIdx.x * blockDim.x + threadIdx.x;
  if (i >= Etot * HH) return;
  const int e = i / HH, h = i % HH;
  const int dn = edst(ei, e, E);
  const float m = funkey(mkey[dn * HH + h]);
  const float v = __expf(s[i] - m);
  ex[i] = v;
  atomicAdd(&den[dn * HH + h], v);
}

__global__ __launch_bounds__(256) void k_edge_aggr(
    const int* __restrict__ ei, const float* __restrict__ ex,
    const float* __restrict__ den, const float* __restrict__ xl,
    float* __restrict__ out, int E, int Etot, int HH, int C)
{
  const int task = blockIdx.x * (blockDim.x >> 5) + (threadIdx.x >> 5);
  if (task >= Etot * HH) return;
  const int e = task / HH, h = task % HH;
  const int lane = threadIdx.x & 31;
  const int sn = esrc(ei, e, E), dn = edst(ei, e, E);
  const float alpha = ex[task] / (den[dn * HH + h] + EPSF);
  const float* pl = xl + ((size_t)sn * HH + h) * C;
  float* po = out + ((size_t)dn * HH + h) * C;
  for (int c = lane; c < C; c += 32) atomicAdd(&po[c], alpha * pl[c]);
}

// mean over heads + bias + relu; also produce bf16 copy for next GEMM
__global__ void k_head_mean(
    const float* __restrict__ agg, const float* __restrict__ bias,
    float* __restrict__ xf, unsigned short* __restrict__ xb,
    int Nn, int HH, int C)
{
  const long i = (long)blockIdx.x * blockDim.x + threadIdx.x;
  if (i >= (long)Nn * C) return;
  const int n = (int)(i / C), c = (int)(i % C);
  float acc = 0.0f;
  for (int h = 0; h < HH; ++h) acc += agg[((size_t)n * HH + h) * C + c];
  acc = acc * (1.0f / (float)HH) + bias[c];
  acc = fmaxf(acc, 0.0f);
  xf[i] = acc;
  xb[i] = f2bf(acc);
}

// ---------------------------------------------------------------------------
// Graph softmax pooling
// ---------------------------------------------------------------------------
__global__ void k_pool_max(const float* __restrict__ g, const int* __restrict__ batch,
                           unsigned int* __restrict__ mk, int Nn) {
  const int i = blockIdx.x * blockDim.x + threadIdx.x;
  if (i < Nn) atomicMax(&mk[batch[i]], fkey(g[i]));
}
__global__ void k_pool_exp(const float* __restrict__ g, const int* __restrict__ batch,
                           const unsigned int* __restrict__ mk, float* __restrict__ ex,
                           float* __restrict__ den, int Nn) {
  const int i = blockIdx.x * blockDim.x + threadIdx.x;
  if (i >= Nn) return;
  const int b = batch[i];
  const float v = __expf(g[i] - funkey(mk[b]));
  ex[i] = v;
  atomicAdd(&den[b], v);
}
__global__ void k_pool_aggr(const float* __restrict__ ex, const float* __restrict__ den,
                            const int* __restrict__ batch, const float* __restrict__ x,
                            float* __restrict__ hbuf, int Nn, int C) {
  const long i = (long)blockIdx.x * blockDim.x + threadIdx.x;
  if (i >= (long)Nn * C) return;
  const int n = (int)(i / C), c = (int)(i % C);
  const int b = batch[n];
  const float w = ex[n] / (den[b] + EPSF);
  atomicAdd(&hbuf[(size_t)b * C + c], w * x[i]);
}

// ---------------------------------------------------------------------------
// Small dense layer (MLP head / gate projection): thread per output element
// ---------------------------------------------------------------------------
__global__ void k_small_gemm(const float* __restrict__ A, const float* __restrict__ W,
                             const float* __restrict__ b, float* __restrict__ C,
                             int M, int K, int Nc, int relu) {
  const int i = blockIdx.x * blockDim.x + threadIdx.x;
  if (i >= M * Nc) return;
  const int m = i / Nc, o = i % Nc;
  float acc = b[o];
  for (int k = 0; k < K; ++k) acc += A[(size_t)m * K + k] * W[(size_t)k * Nc + o];
  C[i] = relu ? fmaxf(acc, 0.0f) : acc;
}

// ---------------------------------------------------------------------------
// Host-side launcher
// ---------------------------------------------------------------------------
extern "C" void kernel_launch(void* const* d_in, const int* in_sizes, int n_in,
                              void* d_out, int out_size, void* d_ws, size_t ws_size,
                              hipStream_t stream) {
  (void)in_sizes; (void)n_in; (void)out_size; (void)ws_size;
  const int N = 5000, E = 50000, G = 8, HH = 4;
  const int Etot = E + N;
  const int dof[3] = {1024, 512, 256};

  const float* x     = (const float*)d_in[0];
  const int*   ei    = (const int*)d_in[1];
  const int*   batch = (const int*)d_in[2];
  const float* Wl[3]  = {(const float*)d_in[3],  (const float*)d_in[9],  (const float*)d_in[15]};
  const float* bl[3]  = {(const float*)d_in[4],  (const float*)d_in[10], (const float*)d_in[16]};
  const float* Wr[3]  = {(const float*)d_in[5],  (const float*)d_in[11], (const float*)d_in[17]};
  const float* br[3]  = {(const float*)d_in[6],  (const float*)d_in[12], (const float*)d_in[18]};
  const float* att[3] = {(const float*)d_in[7],  (const float*)d_in[13], (const float*)d_in[19]};
  const float* bb[3]  = {(const float*)d_in[8],  (const float*)d_in[14], (const float*)d_in[20]};
  const float* Wm[5]  = {(const float*)d_in[21], (const float*)d_in[23], (const float*)d_in[25],
                         (const float*)d_in[27], (const float*)d_in[29]};
  const float* bm[5]  = {(const float*)d_in[22], (const float*)d_in[24], (const float*)d_in[26],
                         (const float*)d_in[28], (const float*)d_in[30]};
  const float* Wg0 = (const float*)d_in[31];
  const float* bg0 = (const float*)d_in[32];
  const float* Wg1 = (const float*)d_in[33];
  const float* bg1 = (const float*)d_in[34];

  // ---- workspace bump allocator ----
  char* wp = (char*)d_ws;
  auto alloc = [&](size_t bytes) -> void* {
    void* p = (void*)wp;
    wp += (bytes + 255) & ~(size_t)255;
    return p;
  };
  unsigned short* xb  = (unsigned short*)alloc((size_t)N * 1536 * 2);
  unsigned short* wbL = (unsigned short*)alloc((size_t)1536 * 4096 * 2);
  unsigned short* wbR = (unsigned short*)alloc((size_t)1536 * 4096 * 2);
  float* xl   = (float*)alloc((size_t)N * 4096 * 4);
  float* xr   = (float*)alloc((size_t)N * 4096 * 4);
  float* agg  = (float*)alloc((size_t)N * 4096 * 4);
  float* xf   = (float*)alloc((size_t)N * 1024 * 4);
  float* sarr = (float*)alloc((size_t)Etot * HH * 4);
  float* exarr= (float*)alloc((size_t)Etot * HH * 4);
  unsigned int* mkey = (unsigned int*)alloc((size_t)N * HH * 4);
  float* den  = (float*)alloc((size_t)N * HH * 4);
  float* gate1= (float*)alloc((size_t)N * 128 * 4);
  float* gs   = (float*)alloc((size_t)N * 4);
  float* gex  = (float*)alloc((size_t)N * 4);
  unsigned int* gmk = (unsigned int*)alloc((size_t)G * 4);
  float* gden = (float*)alloc((size_t)G * 4);
  float* hbuf = (float*)alloc((size_t)G * 256 * 4);
  float* mb0  = (float*)alloc((size_t)G * 128 * 4);
  float* mb1  = (float*)alloc((size_t)G * 128 * 4);

  auto zf = [&](float* p, long n) {
    k_zero_f32<<<(unsigned)((n + 255) / 256), 256, 0, stream>>>(p, n);
  };
  auto zu = [&](unsigned int* p, long n) {
    k_zero_u32<<<(unsigned)((n + 255) / 256), 256, 0, stream>>>(p, n);
  };
  auto cvt = [&](const float* in, unsigned short* out, long n) {
    k_f32_to_bf16<<<(unsigned)((n + 255) / 256), 256, 0, stream>>>(in, out, n);
  };

  // ---- GAT layers ----
  cvt(x, xb, (long)N * 1536);
  int din = 1536;
  for (int i = 0; i < 3; ++i) {
    const int C = dof[i], HC = HH * C;
    const long wn = (long)din * HC;
    cvt(Wl[i], wbL, wn);
    cvt(Wr[i], wbR, wn);

    dim3 gg((N + 127) / 128, HC / 128);
    k_wmma_gemm<<<gg, 256, 0, stream>>>(xb, wbL, bl[i], xl, N, din, HC, 0);
    k_wmma_gemm<<<gg, 256, 0, stream>>>(xb, wbR, br[i], xr, N, din, HC, 0);

    zu(mkey, (long)N * HH);
    zf(den,  (long)N * HH);
    zf(agg,  (long)N * HC);

    const long tasks = (long)Etot * HH;
    k_edge_score<<<(unsigned)((tasks + 7) / 8), 256, 0, stream>>>(
        ei, xl, xr, att[i], sarr, mkey, E, Etot, HH, C);
    k_edge_exp<<<(unsigned)((tasks + 255) / 256), 256, 0, stream>>>(
        ei, sarr, mkey, exarr, den, E, Etot, HH);
    k_edge_aggr<<<(unsigned)((tasks + 7) / 8), 256, 0, stream>>>(
        ei, exarr, den, xl, agg, E, Etot, HH, C);
    k_head_mean<<<(unsigned)(((long)N * C + 255) / 256), 256, 0, stream>>>(
        agg, bb[i], xf, xb, N, HH, C);
    din = C;
  }
  // xf = x3 (N x 256, float), xb = bf16(x3)

  // ---- gate ----
  cvt(Wg0, wbL, 256L * 128);
  dim3 gg2((N + 127) / 128, 1);
  k_wmma_gemm<<<gg2, 256, 0, stream>>>(xb, wbL, bg0, gate1, N, 256, 128, 1);
  k_small_gemm<<<(N + 255) / 256, 256, 0, stream>>>(gate1, Wg1, bg1, gs, N, 128, 1, 0);

  // ---- softmax pooling over graphs ----
  zu(gmk, G);
  zf(gden, G);
  zf(hbuf, (long)G * 256);
  k_pool_max<<<(N + 255) / 256, 256, 0, stream>>>(gs, batch, gmk, N);
  k_pool_exp<<<(N + 255) / 256, 256, 0, stream>>>(gs, batch, gmk, gex, gden, N);
  k_pool_aggr<<<(unsigned)(((long)N * 256 + 255) / 256), 256, 0, stream>>>(
      gex, gden, batch, xf, hbuf, N, 256);

  // ---- MLP head (G = 8 rows; scalar kernels) ----
  k_small_gemm<<<4, 256, 0, stream>>>(hbuf, Wm[0], bm[0], mb0, G, 256, 128, 1);
  k_small_gemm<<<2, 256, 0, stream>>>(mb0,  Wm[1], bm[1], mb1, G, 128,  64, 1);
  k_small_gemm<<<1, 256, 0, stream>>>(mb1,  Wm[2], bm[2], mb0, G,  64,  32, 1);
  k_small_gemm<<<1, 128, 0, stream>>>(mb0,  Wm[3], bm[3], mb1, G,  32,  16, 1);
  k_small_gemm<<<1, 64,  0, stream>>>(mb1,  Wm[4], bm[4], (float*)d_out, G, 16, 1, 0);
}
